// Model_16140487099083
// MI455X (gfx1250) — matrix operand order repaired and checked by reading, not compiled, not run on hardware
//
#include <hip/hip_runtime.h>

typedef _Float16 v16h __attribute__((ext_vector_type(16)));
typedef float    v8f  __attribute__((ext_vector_type(8)));

constexpr int Wn = 32, Cn = 256, Fn = 64, CDn = 32, EDn = 16, En = 8192;
constexpr int Dn = 256, Hn = 128;

__device__ __forceinline__ float geluf(float x) {
  return 0.5f * x * (1.f + erff(x * 0.70710678118654752f));
}
__device__ __forceinline__ float sigmf(float x) { return 1.f / (1.f + expf(-x)); }

__device__ __forceinline__ float blk_sum256(float v, float* sh) {
  int t = threadIdx.x;
  sh[t] = v; __syncthreads();
  for (int s = 128; s > 0; s >>= 1) { if (t < s) sh[t] += sh[t + s]; __syncthreads(); }
  float r = sh[0]; __syncthreads();
  return r;
}
__device__ __forceinline__ float blk_max256(float v, float* sh) {
  int t = threadIdx.x;
  sh[t] = v; __syncthreads();
  for (int s = 128; s > 0; s >>= 1) { if (t < s) sh[t] = fmaxf(sh[t], sh[t + s]); __syncthreads(); }
  float r = sh[0]; __syncthreads();
  return r;
}

// pack 16 f32 (as 4x float4) into a v16h fragment, scaling by s (0 or 1)
__device__ __forceinline__ v16h pack16(float4 x0, float4 x1, float4 x2, float4 x3, float s) {
  v16h r;
  r[0]  = (_Float16)(x0.x * s); r[1]  = (_Float16)(x0.y * s);
  r[2]  = (_Float16)(x0.z * s); r[3]  = (_Float16)(x0.w * s);
  r[4]  = (_Float16)(x1.x * s); r[5]  = (_Float16)(x1.y * s);
  r[6]  = (_Float16)(x1.z * s); r[7]  = (_Float16)(x1.w * s);
  r[8]  = (_Float16)(x2.x * s); r[9]  = (_Float16)(x2.y * s);
  r[10] = (_Float16)(x2.z * s); r[11] = (_Float16)(x2.w * s);
  r[12] = (_Float16)(x3.x * s); r[13] = (_Float16)(x3.y * s);
  r[14] = (_Float16)(x3.z * s); r[15] = (_Float16)(x3.w * s);
  return r;
}
__device__ __forceinline__ float4 ld4(const float* p) { return *(const float4*)p; }

// ---------------------------------------------------------------------------
// WMMA GEMM: Out[M,N] = act( (acc? Out:0) + bias + A[m+shift, :K] @ Wt[n, :K]^T )
// A f32 [MA, lda]; Wt f32 [N, ldw] (torch Linear layout). f32 accumulate via
// v_wmma_f32_16x16x32_f16, operands converted f32->f16 in registers.
// 4 waves/block; each wave owns a 32x32 tile (2 A frags x 2 B frags = 4
// independent WMMAs per K-step); block tile 64x64. K multiple of 32.
// Out-of-range A rows (conv zero-padding via rowShift) are loaded from a
// clamped pointer and zeroed by a 0/1 scale -- fully branch-free inner loop,
// 16 global_load_b128 per 4 WMMAs.
// ---------------------------------------------------------------------------
template <int ACT>  // 0=none, 1=gelu, 2=tanh
__global__ void __launch_bounds__(128)
k_gemm(const float* __restrict__ A, int lda, int MA, int rowShift,
       const float* __restrict__ Wt, int ldw,
       const float* __restrict__ bias,
       float* __restrict__ Out, int ldo,
       int M, int N, int Kd, int accumulate) {
  int lane = threadIdx.x & 31;
  int wv   = threadIdx.x >> 5;
  int tm   = blockIdx.y * 64 + (wv >> 1) * 32;
  int tn   = blockIdx.x * 64 + (wv & 1) * 32;
  int l15  = lane & 15;
  int hi   = lane >> 4;

  // A fragment rows (two 16-row fragments)
  int ar0 = tm + l15 + rowShift;
  int ar1 = ar0 + 16;
  float za0 = (ar0 >= 0 && ar0 < MA) ? 1.f : 0.f;
  float za1 = (ar1 >= 0 && ar1 < MA) ? 1.f : 0.f;
  const float* Ar0 = A + (size_t)(za0 != 0.f ? ar0 : 0) * lda + (hi << 3);
  const float* Ar1 = A + (size_t)(za1 != 0.f ? ar1 : 0) * lda + (hi << 3);
  // B fragment cols (two 16-col fragments)
  int bn0 = tn + l15;
  int bn1 = bn0 + 16;
  const float* Br0 = Wt + (size_t)(bn0 < N ? bn0 : 0) * ldw + (hi << 3);
  const float* Br1 = Wt + (size_t)(bn1 < N ? bn1 : 0) * ldw + (hi << 3);

  v8f acc00 = {0.f,0.f,0.f,0.f,0.f,0.f,0.f,0.f};
  v8f acc01 = acc00, acc10 = acc00, acc11 = acc00;

  for (int k0 = 0; k0 < Kd; k0 += 32) {
    // A layout (16-bit A 16x32): lanes 0-15 K={0..7,16..23}, lanes 16-31 K={8..15,24..31}
    v16h a0 = pack16(ld4(Ar0 + k0), ld4(Ar0 + k0 + 4),
                     ld4(Ar0 + k0 + 16), ld4(Ar0 + k0 + 20), za0);
    v16h a1 = pack16(ld4(Ar1 + k0), ld4(Ar1 + k0 + 4),
                     ld4(Ar1 + k0 + 16), ld4(Ar1 + k0 + 20), za1);
    // B layout (16-bit B 32x16): lanes 0-15 K={0..7,16..23}, lanes 16-31 K={8..15,24..31}
    v16h b0 = pack16(ld4(Br0 + k0), ld4(Br0 + k0 + 4),
                     ld4(Br0 + k0 + 16), ld4(Br0 + k0 + 20), 1.f);
    v16h b1 = pack16(ld4(Br1 + k0), ld4(Br1 + k0 + 4),
                     ld4(Br1 + k0 + 16), ld4(Br1 + k0 + 20), 1.f);
    acc00 = __builtin_amdgcn_wmma_f32_16x16x32_f16(false, a0, false, b0, (short)0, acc00, false, false);
    acc01 = __builtin_amdgcn_wmma_f32_16x16x32_f16(false, a0, false, b1, (short)0, acc01, false, false);
    acc10 = __builtin_amdgcn_wmma_f32_16x16x32_f16(false, a1, false, b0, (short)0, acc10, false, false);
    acc11 = __builtin_amdgcn_wmma_f32_16x16x32_f16(false, a1, false, b1, (short)0, acc11, false, false);
  }

#pragma unroll
  for (int fi = 0; fi < 2; fi++) {
#pragma unroll
    for (int fj = 0; fj < 2; fj++) {
      const v8f& acc = fi ? (fj ? acc11 : acc10) : (fj ? acc01 : acc00);
      int n = tn + fj * 16 + l15;
      if (n < N) {
        float bb = bias ? bias[n] : 0.f;
#pragma unroll
        for (int r = 0; r < 8; r++) {
          int m = tm + fi * 16 + r + (hi << 3);  // C/D: VGPR r -> row r (+8 upper lanes)
          if (m < M) {
            float v = acc[r] + bb;
            if (accumulate) v += Out[(size_t)m * ldo + n];
            if (ACT == 1) v = geluf(v);
            else if (ACT == 2) v = tanhf(v);
            Out[(size_t)m * ldo + n] = v;
          }
        }
      }
    }
  }
}

// ---------------------------- small utility kernels ------------------------
__global__ void k_zero(float* p, int n) {
  int i = blockIdx.x * 256 + threadIdx.x;
  if (i < n) p[i] = 0.f;
}
__global__ void k_mask(const unsigned char* m, float* mf) {
  int i = threadIdx.x;
  mf[i] = m[i] ? 1.f : 0.f;
}
// conv weight (O,I,3) -> wt[t][o][i] so K is contiguous per tap
__global__ void k_convw(const float* __restrict__ cw, float* __restrict__ wt) {
  int i = blockIdx.x * 256 + threadIdx.x;
  if (i >= Dn * Dn * 3) return;
  int t = i % 3; int r = i / 3;
  int din = r % Dn; int o = r / Dn;
  wt[((size_t)t * Dn + o) * Dn + din] = cw[i];
}
// [W,C,D] -> [C,W,D]
__global__ void k_seqT(const float* __restrict__ s, float* __restrict__ d) {
  int i = blockIdx.x * 256 + threadIdx.x;
  if (i >= Wn * Cn * Dn) return;
  int dd = i % Dn; int c = (i / Dn) % Cn; int w = i / (Dn * Cn);
  d[((size_t)c * Wn + w) * Dn + dd] = s[i];
}
// out[row] = dot(A[row, :Kd], w) + b   (Kd <= 256)
__global__ void k_rowdot(const float* __restrict__ A, int lda,
                         const float* __restrict__ w, const float* __restrict__ b,
                         float* __restrict__ out, int Kd) {
  __shared__ float sh[256];
  int row = blockIdx.x, t = threadIdx.x;
  float s = (t < Kd) ? A[(size_t)row * lda + t] * w[t] : 0.f;
  float tot = blk_sum256(s, sh);
  if (t == 0) out[row] = tot + b[0];
}
// one GRU step: torch gate order (r,z,n)
__global__ void k_gru_gate(const float* __restrict__ gi, const float* __restrict__ gh,
                           float* __restrict__ hidden, int wt, int wprev,
                           int dirOff, int first) {
  int idx = blockIdx.x * 256 + threadIdx.x;
  if (idx >= Cn * Hn) return;
  int c = idx / Hn, j = idx % Hn;
  const float* gir = gi + ((size_t)c * Wn + wt) * (3 * Hn);
  const float* ghr = gh + (size_t)c * (3 * Hn);
  float r = sigmf(gir[j] + ghr[j]);
  float z = sigmf(gir[Hn + j] + ghr[Hn + j]);
  float n = tanhf(gir[2 * Hn + j] + r * ghr[2 * Hn + j]);
  float hp = first ? 0.f : hidden[((size_t)c * Wn + wprev) * Dn + dirOff + j];
  hidden[((size_t)c * Wn + wt) * Dn + dirOff + j] = (1.f - z) * n + z * hp;
}
// attention pooling over W per channel
__global__ void k_attnpool(const float* __restrict__ hidden, const float* __restrict__ al,
                           const float* __restrict__ mf, float* __restrict__ ce) {
  __shared__ float aw[Wn];
  int c = blockIdx.x;
  if (threadIdx.x == 0) {
    float mx = -1e30f;
    for (int w = 0; w < Wn; w++) mx = fmaxf(mx, al[c * Wn + w]);
    float s = 0.f;
    for (int w = 0; w < Wn; w++) { float e = expf(al[c * Wn + w] - mx); aw[w] = e; s += e; }
    float inv = 1.f / s;
    for (int w = 0; w < Wn; w++) aw[w] *= inv;
  }
  __syncthreads();
  int d = threadIdx.x;
  float acc = 0.f;
  for (int w = 0; w < Wn; w++) acc += hidden[((size_t)c * Wn + w) * Dn + d] * aw[w];
  ce[c * Dn + d] = acc * mf[c];
}
__global__ void k_wlog(const float* __restrict__ tmp, const float* __restrict__ mf,
                       float* __restrict__ wl) {
  int i = blockIdx.x * 256 + threadIdx.x;
  if (i >= Cn * Wn) return;
  int w = i & (Wn - 1); int c = i >> 5;
  wl[(size_t)w * Cn + c] = tmp[i] * mf[c];
}
// row-wise LayerNorm over D=256, optional residual add, optional gelu
template <int ACT, int RES>
__global__ void k_ln(const float* __restrict__ in, const float* __restrict__ res,
                     const float* __restrict__ g, const float* __restrict__ b,
                     float* __restrict__ out) {
  __shared__ float sh[256];
  int row = blockIdx.x, j = threadIdx.x;
  float v = in[(size_t)row * Dn + j];
  if (RES) v += res[(size_t)row * Dn + j];
  float mean = blk_sum256(v, sh) * (1.f / Dn);
  float d = v - mean;
  float var = blk_sum256(d * d, sh) * (1.f / Dn);
  float o = d * rsqrtf(var + 1e-5f) * g[j] + b[j];
  if (ACT == 1) o = geluf(o);
  out[(size_t)row * Dn + j] = o;
}
__global__ void k_deg(const int* __restrict__ dst, float* __restrict__ deg) {
  int e = blockIdx.x * 256 + threadIdx.x;
  if (e < En) atomicAdd(&deg[dst[e]], 1.f);
}
// fused: er = gelu(LN(edge_attr @ ep_w.T + ep_b)); msg = nh[src] + er; agg[dst] += msg
__global__ void k_edge_scatter(const float* __restrict__ ea, const float* __restrict__ epw,
                               const float* __restrict__ epb, const float* __restrict__ epg,
                               const float* __restrict__ epbt, const float* __restrict__ nh,
                               const int* __restrict__ src, const int* __restrict__ dst,
                               float* __restrict__ agg) {
  __shared__ float sh[256];
  int we = blockIdx.x; int w = we >> 13; int e = we & (En - 1);
  int j = threadIdx.x;
  const float* ar = ea + ((size_t)w * En + e) * EDn;
  float v = epb[j];
#pragma unroll
  for (int k = 0; k < EDn; k++) v += ar[k] * epw[j * EDn + k];
  float mean = blk_sum256(v, sh) * (1.f / Dn);
  float dv = v - mean;
  float var = blk_sum256(dv * dv, sh) * (1.f / Dn);
  float t = geluf(dv * rsqrtf(var + 1e-5f) * epg[j] + epbt[j]);
  int s = src[e], d = dst[e];
  float msg = nh[((size_t)w * Cn + s) * Dn + j] + t;
  atomicAdd(&agg[((size_t)w * Cn + d) * Dn + j], msg);
}
__global__ void k_cat_nh_agg(const float* __restrict__ nh, const float* __restrict__ agg,
                             const float* __restrict__ deg, float* __restrict__ cat) {
  int i = blockIdx.x * 256 + threadIdx.x;
  if (i >= Wn * Cn * 512) return;
  int j = i & 511; int m = i >> 9; int c = m & (Cn - 1);
  cat[i] = (j < 256) ? nh[(size_t)m * Dn + j]
                     : agg[(size_t)m * Dn + (j - 256)] / fmaxf(deg[c], 1.f);
}
__global__ void k_graphemb(const float* __restrict__ nodes, const float* __restrict__ mf,
                           float* __restrict__ ge) {
  int c = blockIdx.x, d = threadIdx.x;
  float s = 0.f;
  for (int w = 0; w < Wn; w++) s += nodes[((size_t)w * Cn + c) * Dn + d];
  ge[c * Dn + d] = s * (1.f / Wn) * mf[c];
}
__global__ void k_concat2(const float* __restrict__ a, const float* __restrict__ b,
                          float* __restrict__ o, int rows) {
  int i = blockIdx.x * 256 + threadIdx.x;
  if (i >= rows * 512) return;
  int j = i & 511; int r = i >> 9;
  o[i] = (j < 256) ? a[r * Dn + j] : b[r * Dn + (j - 256)];
}
__global__ void k_scale_rows(const float* __restrict__ a, const float* __restrict__ mf,
                             float* __restrict__ o) {
  int i = blockIdx.x * 256 + threadIdx.x;
  if (i >= Cn * Dn) return;
  o[i] = a[i] * mf[i >> 8];
}
// attention scores + softmax, 4 heads, 256 keys per block
__global__ void k_scores(const float* __restrict__ qkv, const float* __restrict__ mf,
                         float* __restrict__ att) {
  __shared__ float sh[256];
  int q = blockIdx.x, h = blockIdx.y, k = threadIdx.x;
  const float* qr = qkv + (size_t)q * 768 + h * 64;
  const float* kr = qkv + (size_t)k * 768 + 256 + h * 64;
  float s = 0.f;
#pragma unroll 8
  for (int d = 0; d < 64; d++) s += qr[d] * kr[d];
  s *= 0.125f;  // 1/sqrt(64)
  if (mf[k] == 0.f) s = -1e9f;
  float mx = blk_max256(s, sh);
  float e = expf(s - mx);
  float tot = blk_sum256(e, sh);
  att[((size_t)h * Cn + q) * Cn + k] = e / tot;
}
__global__ void k_attapply(const float* __restrict__ att, const float* __restrict__ qkv,
                           float* __restrict__ o) {
  int q = blockIdx.x, t = threadIdx.x, h = t >> 6, d = t & 63;
  const float* ar = att + ((size_t)h * Cn + q) * Cn;
  float acc = 0.f;
  for (int k = 0; k < Cn; k++) acc += ar[k] * qkv[(size_t)k * 768 + 512 + h * 64 + d];
  o[(size_t)q * Dn + t] = acc;
}
__global__ void k_final(const float* __restrict__ pl0, const float* __restrict__ rl0,
                        const float* __restrict__ mf, float* __restrict__ out) {
  int c = threadIdx.x;
  float m = mf[c];
  float rl = rl0[c] * m;              // run_logits (weight/denom == 1 when masked-in)
  float pl = pl0[c] + 0.5f * rl;
  pl = (m > 0.f) ? pl : -12.f;
  out[c] = pl;
  out[Cn + c] = sigmf(pl) * m;
  out[2 * Cn + c] = sigmf(rl) * m;
}

// ------------------------------ workspace arena ----------------------------
constexpr size_t WCD = (size_t)Wn * Cn * Dn;                 // 2,097,152
constexpr size_t F_S0   = 0;                 // proj -> seqT
constexpr size_t F_S1   = F_S0 + WCD;        // convA -> a1out -> upd
constexpr size_t F_S2   = F_S1 + WCD;        // convB -> nh
constexpr size_t F_S3   = F_S2 + WCD;        // gi_f (3.1M) -> agg
constexpr size_t F_S4   = F_S3 + 3145728;    // gi_b (3.1M) -> u1out
constexpr size_t F_S5   = F_S4 + 3145728;    // hidden -> nodes
constexpr size_t F_S6   = F_S5 + WCD;        // cat [W*C,512]
constexpr size_t F_GH   = F_S6 + 4194304;    // gh [C,384]
constexpr size_t F_H0   = F_GH + 98304;      // h0 zeros [C,H]
constexpr size_t F_WT   = F_H0 + 32768;      // conv wt [3,D,D]
constexpr size_t F_AL   = F_WT + 196608;     // al / tmp wlog [C*W]
constexpr size_t F_CE   = F_AL + 8192;       // channel_emb [C,D]
constexpr size_t F_GE   = F_CE + 65536;      // graph_emb [C,D]
constexpr size_t F_DEG  = F_GE + 65536;      // deg [C]
constexpr size_t F_RCAT = F_DEG + 256;       // [C,512]
constexpr size_t F_REMB = F_RCAT + 131072;   // run_emb [C,D]
constexpr size_t F_RC1  = F_REMB + 65536;    // [C,128]
constexpr size_t F_RL   = F_RC1 + 32768;     // run_logits0 [C]
constexpr size_t F_XA   = F_RL + 256;        // [C,D]
constexpr size_t F_XB   = F_XA + 65536;      // [C,D]
constexpr size_t F_QKV  = F_XB + 65536;      // [C,768]
constexpr size_t F_ATT  = F_QKV + 196608;    // [4,C,C]
constexpr size_t F_AO   = F_ATT + 262144;    // attn out [C,D]
constexpr size_t F_OP   = F_AO + 65536;      // o proj [C,D]
constexpr size_t F_FF1  = F_OP + 65536;      // [C,512]
constexpr size_t F_FF2  = F_FF1 + 131072;    // [C,D]
constexpr size_t F_PH1  = F_FF2 + 65536;     // [C,128]
constexpr size_t F_PL0  = F_PH1 + 32768;     // [C]
constexpr size_t F_MF   = F_PL0 + 256;       // [C]

static void gemm(hipStream_t st, const float* A, int lda, int MA, int shift,
                 const float* Wt, int ldw, const float* bias,
                 float* Out, int ldo, int M, int N, int K, int acc, int act) {
  dim3 g((N + 63) / 64, (M + 63) / 64);
  if (act == 1)      k_gemm<1><<<g, 128, 0, st>>>(A, lda, MA, shift, Wt, ldw, bias, Out, ldo, M, N, K, acc);
  else if (act == 2) k_gemm<2><<<g, 128, 0, st>>>(A, lda, MA, shift, Wt, ldw, bias, Out, ldo, M, N, K, acc);
  else               k_gemm<0><<<g, 128, 0, st>>>(A, lda, MA, shift, Wt, ldw, bias, Out, ldo, M, N, K, acc);
}
static inline dim3 g1(int n) { return dim3((n + 255) / 256); }

extern "C" void kernel_launch(void* const* d_in, const int* in_sizes, int n_in,
                              void* d_out, int out_size, void* d_ws, size_t ws_size,
                              hipStream_t st) {
  (void)in_sizes; (void)n_in; (void)out_size; (void)ws_size;
  const float* x_feat    = (const float*)d_in[0];   // [W,C,F]
  const float* node_conn = (const float*)d_in[1];   // [W,C,CD]
  const float* edge_attr = (const float*)d_in[2];   // [W,E,ED]
  const int*   eidx      = (const int*)d_in[3];     // [2,E]
  const unsigned char* cmask = (const unsigned char*)d_in[4];
  auto P = [&](int i) -> const float* { return (const float*)d_in[5 + i]; };
  float* ws  = (float*)d_ws;
  float* out = (float*)d_out;

  float* proj   = ws + F_S0;  float* seqT  = ws + F_S0;
  float* convA  = ws + F_S1;  float* a1out = ws + F_S1;  float* upd = ws + F_S1;
  float* convB  = ws + F_S2;  float* nh    = ws + F_S2;
  float* gi_f   = ws + F_S3;  float* agg   = ws + F_S3;
  float* gi_b   = ws + F_S4;  float* u1o   = ws + F_S4;
  float* hidden = ws + F_S5;  float* nodes = ws + F_S5;
  float* cat    = ws + F_S6;
  float* gh  = ws + F_GH;   float* h0  = ws + F_H0;  float* wt  = ws + F_WT;
  float* al  = ws + F_AL;   float* ce  = ws + F_CE;  float* ge  = ws + F_GE;
  float* deg = ws + F_DEG;  float* rct = ws + F_RCAT; float* rem = ws + F_REMB;
  float* rc1 = ws + F_RC1;  float* rl0 = ws + F_RL;
  float* xA  = ws + F_XA;   float* xB  = ws + F_XB;  float* qkv = ws + F_QKV;
  float* att = ws + F_ATT;  float* ao  = ws + F_AO;  float* op  = ws + F_OP;
  float* ff1 = ws + F_FF1;  float* ff2 = ws + F_FF2;
  float* ph1 = ws + F_PH1;  float* pl0 = ws + F_PL0; float* mf  = ws + F_MF;

  const int M = Wn * Cn;  // 8192
  k_mask<<<1, 256, 0, st>>>(cmask, mf);

  // ---- temporal encoder ----
  gemm(st, x_feat, Fn, M, 0, P(0), Fn, P(1), proj, Dn, M, Dn, Fn, 0, 0);       // tp
  k_convw<<<g1(Dn * Dn * 3), 256, 0, st>>>(P(2), wt);                           // conv1 wts
  for (int t = 0; t < 3; t++)                                                   // conv1: 3 shifted GEMMs
    gemm(st, proj, Dn, M, (t - 1) * Cn, wt + (size_t)t * Dn * Dn, Dn,
         t == 0 ? P(3) : nullptr, convA, Dn, M, Dn, Dn, t > 0, t == 2 ? 1 : 0);
  k_convw<<<g1(Dn * Dn * 3), 256, 0, st>>>(P(4), wt);                           // conv2 wts
  for (int t = 0; t < 3; t++)
    gemm(st, convA, Dn, M, (t - 1) * Cn, wt + (size_t)t * Dn * Dn, Dn,
         t == 0 ? P(5) : nullptr, convB, Dn, M, Dn, Dn, t > 0, t == 2 ? 1 : 0);
  k_seqT<<<g1(M * Dn), 256, 0, st>>>(convB, seqT);                              // [W,C,D]->[C,W,D]

  gemm(st, seqT, Dn, M, 0, P(6),  Dn, P(8),  gi_f, 3 * Hn, M, 3 * Hn, Dn, 0, 0);
  gemm(st, seqT, Dn, M, 0, P(10), Dn, P(12), gi_b, 3 * Hn, M, 3 * Hn, Dn, 0, 0);
  k_zero<<<g1(Cn * Hn), 256, 0, st>>>(h0, Cn * Hn);
  for (int dir = 0; dir < 2; dir++) {
    const float* whh = dir ? P(11) : P(7);
    const float* bhh = dir ? P(13) : P(9);
    const float* gi  = dir ? gi_b : gi_f;
    int off = dir ? Hn : 0;
    for (int s = 0; s < Wn; s++) {
      int wtm = dir ? (Wn - 1 - s) : s;
      int wpv = dir ? (wtm + 1) : (wtm - 1);
      const float* Ah = (s == 0) ? h0 : (hidden + (size_t)wpv * Dn + off);
      int lda = (s == 0) ? Hn : Wn * Dn;
      gemm(st, Ah, lda, Cn, 0, whh, Hn, bhh, gh, 3 * Hn, Cn, 3 * Hn, Hn, 0, 0);
      k_gru_gate<<<g1(Cn * Hn), 256, 0, st>>>(gi, gh, hidden, wtm, wpv, off, s == 0);
    }
  }
  gemm(st, hidden, Dn, M, 0, P(14), Dn, P(15), a1out, Dn, M, Dn, Dn, 0, 2);     // tanh(a1)
  k_rowdot<<<M, 256, 0, st>>>(a1out, Dn, P(16), P(17), al, Dn);                 // a2 -> al
  k_attnpool<<<Cn, 256, 0, st>>>(hidden, al, mf, ce);                           // channel_emb
  k_rowdot<<<M, 256, 0, st>>>(hidden, Dn, P(18), P(19), al, Dn);                // wh -> tmp
  k_wlog<<<g1(M), 256, 0, st>>>(al, mf, out + 3 * Cn);                          // window_logits

  // ---- graph encoder ----
  gemm(st, node_conn, CDn, M, 0, P(20), CDn, P(21), nh, Dn, M, Dn, CDn, 0, 0);
  k_ln<1, 0><<<M, 256, 0, st>>>(nh, nullptr, P(22), P(23), nh);                 // gelu(LN(nh))
  k_zero<<<g1((int)WCD), 256, 0, st>>>(agg, (int)WCD);
  k_zero<<<1, 256, 0, st>>>(deg, Cn);
  k_deg<<<g1(En), 256, 0, st>>>(eidx + En, deg);
  k_edge_scatter<<<Wn * En, 256, 0, st>>>(edge_attr, P(24), P(25), P(26), P(27),
                                          nh, eidx, eidx + En, agg);            // fused er+gather+scatter
  k_cat_nh_agg<<<g1(M * 512), 256, 0, st>>>(nh, agg, deg, cat);
  gemm(st, cat, 2 * Dn, M, 0, P(28), 2 * Dn, P(29), u1o, Dn, M, Dn, 2 * Dn, 0, 1);
  gemm(st, u1o, Dn, M, 0, P(30), Dn, P(31), upd, Dn, M, Dn, Dn, 0, 0);
  k_ln<0, 1><<<M, 256, 0, st>>>(upd, nh, P(32), P(33), nodes);                  // LN(nh+upd)
  k_graphemb<<<Cn, 256, 0, st>>>(nodes, mf, ge);

  // ---- run fusion + run head ----
  k_concat2<<<g1(Cn * 512), 256, 0, st>>>(ce, ge, rct, Cn);
  gemm(st, rct, 2 * Dn, Cn, 0, P(34), 2 * Dn, P(35), rem, Dn, Cn, Dn, 2 * Dn, 0, 0);
  k_ln<1, 0><<<Cn, 256, 0, st>>>(rem, nullptr, P(36), P(37), rem);              // gelu(LN)
  gemm(st, rem, Dn, Cn, 0, P(38), Dn, P(39), rc1, Hn, Cn, Hn, Dn, 0, 1);
  k_rowdot<<<Cn, 256, 0, st>>>(rc1, Hn, P(40), P(41), rl0, Hn);                 // run_logits (pre-mask)
  k_scale_rows<<<g1(Cn * Dn), 256, 0, st>>>(rem, mf, xA);                       // patient_emb

  // ---- comparator: 2 CoTAR blocks ----
  for (int i = 0; i < 2; i++) {
    int b = 42 + 12 * i;
    gemm(st, xA, Dn, Cn, 0, P(b + 0), Dn, P(b + 1), qkv, 3 * Dn, Cn, 3 * Dn, Dn, 0, 0);
    k_scores<<<dim3(Cn, 4), 256, 0, st>>>(qkv, mf, att);
    k_attapply<<<Cn, 256, 0, st>>>(att, qkv, ao);
    gemm(st, ao, Dn, Cn, 0, P(b + 2), Dn, P(b + 3), op, Dn, Cn, Dn, Dn, 0, 0);
    k_ln<0, 1><<<Cn, 256, 0, st>>>(op, xA, P(b + 4), P(b + 5), xB);             // LN(x+o)
    gemm(st, xB, Dn, Cn, 0, P(b + 6), Dn, P(b + 7), ff1, 2 * Dn, Cn, 2 * Dn, Dn, 0, 1);
    gemm(st, ff1, 2 * Dn, Cn, 0, P(b + 8), 2 * Dn, P(b + 9), ff2, Dn, Cn, Dn, 2 * Dn, 0, 0);
    k_ln<0, 1><<<Cn, 256, 0, st>>>(ff2, xB, P(b + 10), P(b + 11), xA);          // LN(x+ff)
  }

  // ---- patient head ----
  gemm(st, xA, Dn, Cn, 0, P(66), Dn, P(67), ph1, Hn, Cn, Hn, Dn, 0, 1);
  k_rowdot<<<Cn, 256, 0, st>>>(ph1, Hn, P(68), P(69), pl0, Hn);
  k_final<<<1, 256, 0, st>>>(pl0, rl0, mf, out);
}
